// KnowledgeGraphGNN_773094114150
// MI455X (gfx1250) — matrix operand order repaired and checked by reading, compiled, not run on hardware
//
#include <hip/hip_runtime.h>

// ---------------------------------------------------------------------------
// GCN forward on MI455X (gfx1250, wave32, WMMA).
// adj @ support dominates: 256 MiB fp32 adjacency streamed from HBM per layer
// (does not fit 192MB L2) -> memory bound at 23.3 TB/s (~33us for 3 layers).
// bf16 WMMA keeps the 51.5 GFLOP of matrix math far under the compute roof;
// adj (binary) -> bf16 is lossless, accumulation stays fp32.
// Aggregate kernel stages the support panel through LDS with double-buffered
// async global->LDS copies (ASYNCcnt); B-fragment loads / WMMAs are batched
// per K-step for back-to-back WMMA issue.
// ---------------------------------------------------------------------------

typedef __bf16 bf16_t;
typedef __attribute__((ext_vector_type(16))) __bf16 v16bf;
typedef __attribute__((ext_vector_type(8)))  __bf16 v8bf;
typedef __attribute__((ext_vector_type(8)))  float  v8f;
typedef __attribute__((ext_vector_type(4)))  int    v4i;

// address-space-qualified int32x4 for the async-to-LDS builtin
// (param types per hipcc diagnostic: int4 addrspace(1)* / addrspace(3)*)
typedef __attribute__((address_space(1))) v4i gv4i;
typedef __attribute__((address_space(3))) v4i lv4i;

#define GCN_N     8192
#define GCN_DH    128
#define GCN_DOUT  64

#if __has_builtin(__builtin_amdgcn_global_load_async_to_lds_b128) && \
    __has_builtin(__builtin_amdgcn_s_wait_asynccnt)
#define GCN_ASYNC_LDS 1
#endif

// --- WMMA helpers -----------------------------------------------------------

__device__ __forceinline__ v8f wmma_bf16(v16bf a, v16bf b, v8f c) {
    // D(f32 16x16) = A(bf16 16x32) * B(bf16 32x16) + C
    return __builtin_amdgcn_wmma_f32_16x16x32_bf16(
        /*neg_a=*/false, a, /*neg_b=*/false, b,
        /*c_mod=*/(short)0, c, /*reuse_a=*/false, /*reuse_b=*/false);
}

// A fragment from a row-major bf16 matrix [*, lda].
// Lane l: M = l&15 ; K runs [kb, kb+16) contiguous -> v0..7,
// kb = k0 + (l>>4)*16  (matches B fragment K order).
__device__ __forceinline__ v16bf load_a_bf16(const bf16_t* __restrict__ A,
                                             int lda, int row_base, int k0, int lane) {
    const int row = row_base + (lane & 15);
    const int kb  = k0 + ((lane >> 4) << 4);
    const bf16_t* p = A + (size_t)row * lda + kb;
    v8bf lo = *(const v8bf*)(p);
    v8bf hi = *(const v8bf*)(p + 8);
    v16bf a;
#pragma unroll
    for (int e = 0; e < 8; ++e) { a[e] = lo[e]; a[8 + e] = hi[e]; }
    return a;
}

// Same A fragment but sourced from row-major fp32 (adjacency), converted
// in-register to bf16 (exact for binary adj).
__device__ __forceinline__ v16bf load_a_f32cvt(const float* __restrict__ A,
                                               int lda, int row_base, int k0, int lane) {
    const int row = row_base + (lane & 15);
    const int kb  = k0 + ((lane >> 4) << 4);
    const float* p = A + (size_t)row * lda + kb;
    v8f f0 = *(const v8f*)(p);
    v8f f1 = *(const v8f*)(p + 8);
    v16bf a;
#pragma unroll
    for (int e = 0; e < 8; ++e) { a[e] = (bf16_t)f0[e]; a[8 + e] = (bf16_t)f1[e]; }
    return a;
}

// B fragment from a TRANSPOSED bf16 operand BT[n][k] (row stride ldb in k).
// Lane l: N = col_base + (l&15) ; K = k0 + (l>>4)*16 .. +16 contiguous
// (this code's contiguous K order, same as the A fragment; not the ISA table order).
__device__ __forceinline__ v16bf load_b_bt(const bf16_t* __restrict__ BT,
                                           int ldb, int col_base, int k0, int lane) {
    const int col = col_base + (lane & 15);
    const int kb  = k0 + ((lane >> 4) << 4);
    const bf16_t* p = BT + (size_t)col * ldb + kb;
    v8bf lo = *(const v8bf*)(p);
    v8bf hi = *(const v8bf*)(p + 8);
    v16bf b;
#pragma unroll
    for (int e = 0; e < 8; ++e) { b[e] = lo[e]; b[8 + e] = hi[e]; }
    return b;
}

// 16-byte global -> LDS copy: async engine (ASYNCcnt) when available,
// otherwise synchronous register-staged copy (barrier provides visibility).
__device__ __forceinline__ void cp16_g2lds(bf16_t* lds_dst, const bf16_t* gsrc) {
#ifdef GCN_ASYNC_LDS
    __builtin_amdgcn_global_load_async_to_lds_b128(
        (gv4i*)gsrc, (lv4i*)lds_dst, /*offset=*/0, /*cpol=*/0);
#else
    *(v8bf*)lds_dst = *(const v8bf*)gsrc;
#endif
}

__device__ __forceinline__ void wait_async_all() {
#ifdef GCN_ASYNC_LDS
    __builtin_amdgcn_s_wait_asynccnt(0);
#endif
}

// --- small elementwise / prep kernels --------------------------------------

__global__ void gcn_cvt_f32_bf16(const float* __restrict__ src,
                                 bf16_t* __restrict__ dst, int n) {
    int i = blockIdx.x * blockDim.x + threadIdx.x;
    if (i < n) dst[i] = (bf16_t)src[i];
}

// dst[j*K + k] = (bf16) src[k*J + j]   (transpose [K,J] -> [J,K])
__global__ void gcn_transpose_bf16(const float* __restrict__ src,
                                   bf16_t* __restrict__ dst, int K, int J) {
    int idx = blockIdx.x * blockDim.x + threadIdx.x;
    if (idx < K * J) {
        int k = idx / J, j = idx % J;
        dst[j * K + k] = (bf16_t)src[idx];
    }
}

// --- supportT = (h @ W)^T,  h:[N,128] bf16, WT:[128,128] bf16 (W transposed)
// 8 waves/WG, wave = 16 rows x 128 cols (8 WMMA tiles), K loop 128.
__global__ __launch_bounds__(256) void gcn_support_kernel(
    const bf16_t* __restrict__ h, const bf16_t* __restrict__ WT,
    bf16_t* __restrict__ supportT) {
    const int lane = threadIdx.x & 31;
    const int wave = threadIdx.x >> 5;
    const int row_base = blockIdx.x * 128 + wave * 16;

    v8f acc[8] = {};
#pragma unroll
    for (int k0 = 0; k0 < GCN_DH; k0 += 32) {
        v16bf a = load_a_bf16(h, GCN_DH, row_base, k0, lane);
        v16bf bfrag[8];
#pragma unroll
        for (int t = 0; t < 8; ++t) bfrag[t] = load_b_bt(WT, GCN_DH, t * 16, k0, lane);
#pragma unroll
        for (int t = 0; t < 8; ++t) acc[t] = wmma_bf16(a, bfrag[t], acc[t]);
    }
    // Store transposed: D lane layout (N = lane&15, M = (lane>>4)*8 + r) makes
    // the 8 per-lane values contiguous along M of supportT[n][m].
    const int m_base = row_base + ((lane >> 4) << 3);
#pragma unroll
    for (int t = 0; t < 8; ++t) {
        const int n = t * 16 + (lane & 15);
        v8bf o;
#pragma unroll
        for (int r = 0; r < 8; ++r) o[r] = (bf16_t)acc[t][r];
        *(v8bf*)(supportT + (size_t)n * GCN_N + m_base) = o;
    }
}

// --- z = relu(adj @ support + bias), adj fp32 [N,N] streamed from HBM.
// B panel staged through LDS: double-buffered 64-wide K slabs, async prefetch
// of slab k+64 overlaps the 16 WMMAs consuming slab k. Rows padded to 144 B
// so B-fragment ds_load_b128 reads are spread across banks.
#define GCN_KSTEP   64
#define GCN_LDSROW  72   /* bf16 elements per padded LDS row (144 bytes) */

__global__ __launch_bounds__(256) void gcn_aggregate_kernel(
    const float* __restrict__ adj, const bf16_t* __restrict__ supportT,
    const float* __restrict__ bias, float* __restrict__ z) {
    __shared__ bf16_t smem[2 * GCN_DH * GCN_LDSROW];  // 2 x 18 KiB

    const int tid  = threadIdx.x;
    const int lane = tid & 31;
    const int wave = tid >> 5;
    const int row_base = blockIdx.x * 128 + wave * 16;

    // Stage supportT[n][k0 .. k0+64) into smem buffer `buf`:
    // 128 rows x 128 B = 1024 16-byte chunks, 4 per thread.
    auto stage = [&](int buf, int k0) {
        bf16_t* dstbase = smem + buf * (GCN_DH * GCN_LDSROW);
#pragma unroll
        for (int it = 0; it < 4; ++it) {
            const int c = tid + it * 256;   // chunk id 0..1023
            const int n = c >> 3;           // support column (LDS row)
            const int j = c & 7;            // 16B chunk within row
            cp16_g2lds(dstbase + n * GCN_LDSROW + j * 8,
                       supportT + (size_t)n * GCN_N + k0 + j * 8);
        }
    };

    stage(0, 0);

    v8f acc[8] = {};
    for (int k0 = 0; k0 < GCN_N; k0 += GCN_KSTEP) {
        const int buf = (k0 >> 6) & 1;
        wait_async_all();       // my staged chunks for `buf` have landed
        __syncthreads();        // everyone's chunks landed; buf^1 free to reuse
        if (k0 + GCN_KSTEP < GCN_N) stage(buf ^ 1, k0 + GCN_KSTEP);

        const bf16_t* bbase = smem + buf * (GCN_DH * GCN_LDSROW);
#pragma unroll
        for (int kc = 0; kc < GCN_KSTEP; kc += 32) {
            v16bf a = load_a_f32cvt(adj, GCN_N, row_base, k0 + kc, lane);
            v16bf bfrag[8];
#pragma unroll
            for (int t = 0; t < 8; ++t) {
                const int col  = t * 16 + (lane & 15);
                const int kloc = kc + ((lane >> 4) << 4);
                const bf16_t* p = bbase + col * GCN_LDSROW + kloc;
                v8bf lo = *(const v8bf*)(p);
                v8bf hi = *(const v8bf*)(p + 8);
#pragma unroll
                for (int e = 0; e < 8; ++e) { bfrag[t][e] = lo[e]; bfrag[t][8 + e] = hi[e]; }
            }
#pragma unroll
            for (int t = 0; t < 8; ++t) acc[t] = wmma_bf16(a, bfrag[t], acc[t]);
        }
    }

    const int m_base = row_base + ((lane >> 4) << 3);
#pragma unroll
    for (int t = 0; t < 8; ++t) {
        const int n = t * 16 + (lane & 15);
        const float bn = bias[n];
#pragma unroll
        for (int r = 0; r < 8; ++r) {
            float v = acc[t][r] + bn;
            z[(size_t)(m_base + r) * GCN_DH + n] = v > 0.f ? v : 0.f;
        }
    }
}

// --- per-column batch stats (deterministic tree reduction, 1 WG per column)
__global__ __launch_bounds__(256) void gcn_colstats_kernel(
    const float* __restrict__ z, float* __restrict__ mean, float* __restrict__ rstd) {
    __shared__ float ssum[256];
    __shared__ float ssq[256];
    const int j = blockIdx.x;
    const int t = threadIdx.x;
    float s = 0.f, s2 = 0.f;
    for (int i = t; i < GCN_N; i += 256) {
        float v = z[(size_t)i * GCN_DH + j];
        s += v; s2 += v * v;
    }
    ssum[t] = s; ssq[t] = s2;
    __syncthreads();
    for (int off = 128; off > 0; off >>= 1) {
        if (t < off) { ssum[t] += ssum[t + off]; ssq[t] += ssq[t + off]; }
        __syncthreads();
    }
    if (t == 0) {
        float mu  = ssum[0] * (1.f / GCN_N);
        float var = ssq[0] * (1.f / GCN_N) - mu * mu;
        mean[j] = mu;
        rstd[j] = rsqrtf(var + 1e-5f);
    }
}

// --- h = bf16( (z - mu) * rstd * gamma + beta )
__global__ void gcn_bnapply_kernel(const float* __restrict__ z,
                                   const float* __restrict__ mean,
                                   const float* __restrict__ rstd,
                                   const float* __restrict__ g,
                                   const float* __restrict__ be,
                                   bf16_t* __restrict__ hout) {
    int idx = blockIdx.x * blockDim.x + threadIdx.x;  // N*DH threads
    int j = idx & (GCN_DH - 1);
    float v = (z[idx] - mean[j]) * rstd[j] * g[j] + be[j];
    hout[idx] = (bf16_t)v;
}

// --- out = h @ Wout + bout   (N x 64 fp32), WoutT:[64,128] bf16
__global__ __launch_bounds__(256) void gcn_out_kernel(
    const bf16_t* __restrict__ h, const bf16_t* __restrict__ WoutT,
    const float* __restrict__ bout, float* __restrict__ out) {
    const int lane = threadIdx.x & 31;
    const int wave = threadIdx.x >> 5;
    const int row_base = blockIdx.x * 128 + wave * 16;

    v8f acc[4] = {};
#pragma unroll
    for (int k0 = 0; k0 < GCN_DH; k0 += 32) {
        v16bf a = load_a_bf16(h, GCN_DH, row_base, k0, lane);
        v16bf bfrag[4];
#pragma unroll
        for (int t = 0; t < 4; ++t) bfrag[t] = load_b_bt(WoutT, GCN_DH, t * 16, k0, lane);
#pragma unroll
        for (int t = 0; t < 4; ++t) acc[t] = wmma_bf16(a, bfrag[t], acc[t]);
    }
    const int m_base = row_base + ((lane >> 4) << 3);
#pragma unroll
    for (int t = 0; t < 4; ++t) {
        const int n = t * 16 + (lane & 15);
        const float bn = bout[n];
#pragma unroll
        for (int r = 0; r < 8; ++r)
            out[(size_t)(m_base + r) * GCN_DOUT + n] = acc[t][r] + bn;
    }
}

// ---------------------------------------------------------------------------

extern "C" void kernel_launch(void* const* d_in, const int* in_sizes, int n_in,
                              void* d_out, int out_size, void* d_ws, size_t ws_size,
                              hipStream_t stream) {
    const float* x    = (const float*)d_in[0];
    const float* adj  = (const float*)d_in[1];
    const float* W[3]  = {(const float*)d_in[2], (const float*)d_in[6],  (const float*)d_in[10]};
    const float* b[3]  = {(const float*)d_in[3], (const float*)d_in[7],  (const float*)d_in[11]};
    const float* g[3]  = {(const float*)d_in[4], (const float*)d_in[8],  (const float*)d_in[12]};
    const float* be[3] = {(const float*)d_in[5], (const float*)d_in[9],  (const float*)d_in[13]};
    const float* Wout = (const float*)d_in[14];
    const float* bout = (const float*)d_in[15];
    float* out = (float*)d_out;

    // workspace carve-up (~8.3 MiB)
    char* ws = (char*)d_ws;
    bf16_t* hbuf     = (bf16_t*)(ws);                              // 2 MiB  [N,128] bf16
    bf16_t* supportT = (bf16_t*)(ws + (2u << 20));                 // 2 MiB  [128,N] bf16
    float*  z        = (float*) (ws + (4u << 20));                 // 4 MiB  [N,128] f32
    bf16_t* WT       = (bf16_t*)(ws + (8u << 20));                 // 32 KiB [128,128] bf16
    bf16_t* WoutT    = (bf16_t*)(ws + (8u << 20) + (64u << 10));   // 16 KiB [64,128] bf16
    float*  mean     = (float*) (ws + (8u << 20) + (128u << 10));  // 512 B
    float*  rstd     = mean + 128;

    const int NWG = GCN_N / 128;  // 64 workgroups of 8 waves (16 rows each)

    gcn_cvt_f32_bf16<<<(GCN_N * GCN_DH + 255) / 256, 256, 0, stream>>>(x, hbuf, GCN_N * GCN_DH);

    for (int i = 0; i < 3; ++i) {
        gcn_transpose_bf16<<<(GCN_DH * GCN_DH + 255) / 256, 256, 0, stream>>>(W[i], WT, GCN_DH, GCN_DH);
        gcn_support_kernel<<<NWG, 256, 0, stream>>>(hbuf, WT, supportT);
        gcn_aggregate_kernel<<<NWG, 256, 0, stream>>>(adj, supportT, b[i], z);
        gcn_colstats_kernel<<<GCN_DH, 256, 0, stream>>>(z, mean, rstd);
        gcn_bnapply_kernel<<<(GCN_N * GCN_DH) / 256, 256, 0, stream>>>(z, mean, rstd, g[i], be[i], hbuf);
    }

    gcn_transpose_bf16<<<(GCN_DH * GCN_DOUT + 255) / 256, 256, 0, stream>>>(Wout, WoutT, GCN_DH, GCN_DOUT);
    gcn_out_kernel<<<NWG, 256, 0, stream>>>(hbuf, WoutT, bout, out);
}
